// MultiHead_22892175687949
// MI455X (gfx1250) — matrix operand order repaired and checked by reading, not compiled, not run on hardware
//
#include <hip/hip_runtime.h>
#include <hip/hip_bf16.h>
#include <stdint.h>

// Problem constants (match reference)
#define NUM_HEADS 16
#define DIM_MODEL 1024
#define DK        64
#define BB        2
#define SSEQ      2048
#define MTOT      (BB * SSEQ)   // 4096 rows for all projections

typedef __attribute__((ext_vector_type(16))) __bf16 v16bf;
typedef __attribute__((ext_vector_type(8)))  float  v8f;
typedef __attribute__((ext_vector_type(4)))  int    v4i;

#define LDS_AS __attribute__((address_space(3)))
#define GLB_AS __attribute__((address_space(1)))

#if __has_builtin(__builtin_amdgcn_global_load_async_to_lds_b128)
#define HAVE_ASYNC 1
#else
#define HAVE_ASYNC 0
#endif

union FragA { v16bf bf; uint4 q[2]; unsigned short h[16]; };
union AccU  { v8f f; float e[8]; };

__device__ __forceinline__ unsigned short f32_to_bf16_rte(float f) {
  uint32_t u = __builtin_bit_cast(uint32_t, f);
  uint32_t r = u + 0x7FFFu + ((u >> 16) & 1u);
  return (unsigned short)(r >> 16);
}

__device__ __forceinline__ v8f wmma_bf16(v16bf a, v16bf b, v8f c) {
  // 8 args: (neg_a, A, neg_b, B, c_mod, C, reuse_a, reuse_b)
  return __builtin_amdgcn_wmma_f32_16x16x32_bf16(false, a, false, b, (short)0, c,
                                                 false, false);
}

// 16-byte global -> LDS copy: async (ASYNCcnt) when available, else via VGPRs.
__device__ __forceinline__ void cp16(unsigned short* ldst,
                                     const unsigned short* gsrc) {
#if HAVE_ASYNC
  __builtin_amdgcn_global_load_async_to_lds_b128(
      (GLB_AS v4i*)(void*)gsrc, (LDS_AS v4i*)ldst, 0, 0);
#else
  *(uint4*)ldst = *(const uint4*)gsrc;
#endif
}

__device__ __forceinline__ void wait_async() {
#if HAVE_ASYNC
#if __has_builtin(__builtin_amdgcn_s_wait_asynccnt)
  __builtin_amdgcn_s_wait_asynccnt(0);
#else
  asm volatile("s_wait_asynccnt 0x0" ::: "memory");
#endif
#endif
}

// Cross-lane xor-swap within 16-lane halves via DS_SWIZZLE_B32 (imm pattern,
// group-of-32 mode: offset = xor<<10 | and 0x1f). No index VGPR needed.
template <int PAT>
__device__ __forceinline__ float swzf(float v) {
  return __builtin_bit_cast(
      float, __builtin_amdgcn_ds_swizzle(__builtin_bit_cast(int, v), PAT));
}
__device__ __forceinline__ float redmax16(float v) {
  v = fmaxf(v, swzf<0x041F>(v));
  v = fmaxf(v, swzf<0x081F>(v));
  v = fmaxf(v, swzf<0x101F>(v));
  v = fmaxf(v, swzf<0x201F>(v));
  return v;
}
__device__ __forceinline__ float redsum16(float v) {
  v += swzf<0x041F>(v);
  v += swzf<0x081F>(v);
  v += swzf<0x101F>(v);
  v += swzf<0x201F>(v);
  return v;
}

// ---------------------------------------------------------------------------
// fp32 -> bf16 conversion, 4 elements/thread/step
// ---------------------------------------------------------------------------
__global__ void cvt_f32_bf16_v4(const float4* __restrict__ in,
                                uint2* __restrict__ out, int n4) {
  int i = blockIdx.x * blockDim.x + threadIdx.x;
  int stride = gridDim.x * blockDim.x;
  for (; i < n4; i += stride) {
    float4 f = in[i];
    uint2 o;
    o.x = (unsigned)f32_to_bf16_rte(f.x) | ((unsigned)f32_to_bf16_rte(f.y) << 16);
    o.y = (unsigned)f32_to_bf16_rte(f.z) | ((unsigned)f32_to_bf16_rte(f.w) << 16);
    out[i] = o;
  }
}

// ---------------------------------------------------------------------------
// Tiled GEMM: Y[M,N] = A[M,K] @ W[N,K]^T + bias[N]
// MODE 0: store bf16 to head-split [B,H,S,dk], scaled by `scale`
// MODE 1: store fp32 row-major [M,N]
// 256 threads (8 waves). Block tile 256x128, wave tile 64x64.
// Double-buffered LDS, async global->LDS staging (one barrier per K-step).
// ---------------------------------------------------------------------------
#define GBM 256
#define GBN 128
#define GBK 32
#define LDP 40   // padded pitch in halves (80B rows: conflict-free 16-row reads)

template <int MODE>
__global__ __launch_bounds__(256) void gemm_bf16_wmma(
    const unsigned short* __restrict__ A,   // [M,K] bf16
    const unsigned short* __restrict__ W,   // [N,K] bf16
    const float* __restrict__ bias,         // [N] fp32
    void* __restrict__ outp, int M, int N, int K, float scale) {
  __shared__ alignas(16) unsigned short Alds[2][GBM * LDP];
  __shared__ alignas(16) unsigned short Blds[2][GBN * LDP];

  const int tid  = threadIdx.x;
  const int lane = tid & 31;
  const int wave = tid >> 5;
  const int r16  = lane & 15;
  const int hi   = lane >> 4;

  const int mBase = blockIdx.y * GBM;
  const int nBase = blockIdx.x * GBN;
  const int wm = wave & 3;   // 4 M-slices of 64
  const int wn = wave >> 2;  // 2 N-slices of 64

  AccU acc[4][4];
#pragma unroll
  for (int i = 0; i < 4; i++)
#pragma unroll
    for (int j = 0; j < 4; j++)
#pragma unroll
      for (int e = 0; e < 8; e++) acc[i][j].e[e] = 0.0f;

  auto stageAB = [&](int buf, int kb) {
    // A tile: 256 rows x 32 halves = 1024 16B-chunks, 4 per thread
#pragma unroll
    for (int c2 = 0; c2 < 4; c2++) {
      int id = tid + c2 * 256;
      int row = id >> 2, c = id & 3;
      cp16(&Alds[buf][row * LDP + c * 8],
           A + (size_t)(mBase + row) * K + kb + c * 8);
    }
    // B tile: 128 rows x 32 halves = 512 chunks, 2 per thread
#pragma unroll
    for (int c2 = 0; c2 < 2; c2++) {
      int id = tid + c2 * 256;
      int row = id >> 2, c = id & 3;
      cp16(&Blds[buf][row * LDP + c * 8],
           W + (size_t)(nBase + row) * K + kb + c * 8);
    }
  };

  stageAB(0, 0);
  int cur = 0;
  for (int kb = 0; kb < K; kb += GBK) {
    wait_async();        // my async copies into buf `cur` are done
    __syncthreads();     // everyone's are done; reads of buf `cur^1` retired
    if (kb + GBK < K) stageAB(cur ^ 1, kb + GBK);

    FragA afr[4], bfr[4];
#pragma unroll
    for (int i = 0; i < 4; i++) {
      // A 16x32: lanes 0-15 -> K {0-7,16-23}; lanes 16-31 -> {8-15,24-31}
      const unsigned short* p =
          &Alds[cur][(wm * 64 + i * 16 + r16) * LDP + hi * 8];
      afr[i].q[0] = *(const uint4*)p;
      afr[i].q[1] = *(const uint4*)(p + 16);
    }
#pragma unroll
    for (int j = 0; j < 4; j++) {
      // B 32x16: lane = N col; lanes 0-15 -> K {0-7,16-23}; lanes 16-31 -> {8-15,24-31}
      const unsigned short* p =
          &Blds[cur][(wn * 64 + j * 16 + r16) * LDP + hi * 8];
      bfr[j].q[0] = *(const uint4*)p;
      bfr[j].q[1] = *(const uint4*)(p + 16);
    }
#pragma unroll
    for (int i = 0; i < 4; i++)
#pragma unroll
      for (int j = 0; j < 4; j++)
        acc[i][j].f = wmma_bf16(afr[i].bf, bfr[j].bf, acc[i][j].f);
    cur ^= 1;
  }

  // Epilogue (C/D layout: VGPR e -> M = e + hi*8, lane&15 -> N)
#pragma unroll
  for (int j = 0; j < 4; j++) {
    int col = nBase + wn * 64 + j * 16 + r16;
    float bv = bias[col];
#pragma unroll
    for (int i = 0; i < 4; i++) {
      int row0 = mBase + wm * 64 + i * 16 + hi * 8;
#pragma unroll
      for (int e = 0; e < 8; e++) {
        float v = acc[i][j].e[e] + bv;
        int row = row0 + e;
        if (MODE == 0) {
          int b = row >> 11;
          int s = row & (SSEQ - 1);
          int h = col >> 6;
          int d = col & (DK - 1);
          ((unsigned short*)outp)[(((size_t)(b * NUM_HEADS + h) * SSEQ + s) << 6) + d] =
              f32_to_bf16_rte(v * scale);
        } else {
          ((float*)outp)[(size_t)row * N + col] = v;
        }
      }
    }
  }
}

// ---------------------------------------------------------------------------
// Flash attention: softmax(Q K^T) V per (b,h). 256 threads (8 waves), each
// wave owns 16 q-rows. 64 keys per iteration: 16 WMMAs per softmax pass.
// K tile staged via async copies; V staged transposed via scalar scatter.
// ---------------------------------------------------------------------------
#define QB 128
#define KT 64
#define KP 72   // pitches in halves: 144B rows -> 16B aligned, conflict-free
#define VP 72
#define PP 72

__global__ __launch_bounds__(256) void attn_fwd(
    const unsigned short* __restrict__ Qh,  // [B,H,S,dk] bf16 (pre-scaled)
    const unsigned short* __restrict__ Kh,  // [B,H,S,dk] bf16
    const unsigned short* __restrict__ Vh,  // [B,H,S,dk] bf16
    unsigned short* __restrict__ Out) {     // [B,S,D] bf16, heads merged
  __shared__ alignas(16) unsigned short Klds[KT * KP];      // [key][d]
  __shared__ alignas(16) unsigned short VTlds[DK * VP];     // [d][key]
  __shared__ alignas(16) unsigned short Plds[8 * 16 * PP];  // per-wave P

  const int tid  = threadIdx.x;
  const int lane = tid & 31;
  const int wave = tid >> 5;
  const int r16  = lane & 15;
  const int hi   = lane >> 4;

  const int b = blockIdx.z, h = blockIdx.y;
  const int qbase = blockIdx.x * QB + wave * 16;
  const size_t headOff = (size_t)(b * NUM_HEADS + h) * SSEQ * DK;

  FragA qf[2];  // 16 q-rows x dk=64, two A-frags (d 0..31, 32..63)
  {
    const unsigned short* qp = Qh + headOff + (size_t)(qbase + r16) * DK;
    qf[0].q[0] = *(const uint4*)(qp + hi * 8);
    qf[0].q[1] = *(const uint4*)(qp + hi * 8 + 16);
    qf[1].q[0] = *(const uint4*)(qp + 32 + hi * 8);
    qf[1].q[1] = *(const uint4*)(qp + 32 + hi * 8 + 16);
  }

  AccU acc[4];
#pragma unroll
  for (int n = 0; n < 4; n++)
#pragma unroll
    for (int e = 0; e < 8; e++) acc[n].e[e] = 0.0f;
  float mrow[8], lrow[8];
#pragma unroll
  for (int e = 0; e < 8; e++) { mrow[e] = -1e30f; lrow[e] = 0.0f; }

  unsigned short* myP = &Plds[wave * 16 * PP];

  for (int kb = 0; kb < SSEQ; kb += KT) {
    // ---- stage K (async) : 64 rows x 64 halves = 512 chunks, 2/thread
#pragma unroll
    for (int c2 = 0; c2 < 2; c2++) {
      int id = tid + c2 * 256;
      int key = id >> 3, c = id & 7;
      cp16(&Klds[key * KP + c * 8],
           Kh + headOff + (size_t)(kb + key) * DK + c * 8);
    }
    // ---- stage V transposed: [d][key]
#pragma unroll
    for (int c2 = 0; c2 < 2; c2++) {
      int id = tid + c2 * 256;
      int key = id >> 3, c = id & 7;
      union { uint4 v; unsigned short s[8]; } u;
      u.v = *(const uint4*)(Vh + headOff + (size_t)(kb + key) * DK + c * 8);
#pragma unroll
      for (int e = 0; e < 8; e++) VTlds[(c * 8 + e) * VP + key] = u.s[e];
    }
    wait_async();
    __syncthreads();

    // ---- scores: 4 tiles of 16x16 over keys kb+16t, K-dim = dk = 64
    AccU s[4];
#pragma unroll
    for (int t = 0; t < 4; t++) {
#pragma unroll
      for (int e = 0; e < 8; e++) s[t].e[e] = 0.0f;
#pragma unroll
      for (int dd = 0; dd < 2; dd++) {
        FragA kf;  // B 32x16: lane = key col, d {hi*8+0..7, 16+hi*8+0..7}
        const unsigned short* p = &Klds[(t * 16 + r16) * KP + dd * 32 + hi * 8];
        kf.q[0] = *(const uint4*)p;
        kf.q[1] = *(const uint4*)(p + 16);
        s[t].f = wmma_bf16(qf[dd].bf, kf.bf, s[t].f);
      }
    }

    // ---- online softmax: one reduction pass per 64 keys
    float fac[8];
#pragma unroll
    for (int e = 0; e < 8; e++) {
      float tmax = fmaxf(fmaxf(s[0].e[e], s[1].e[e]),
                         fmaxf(s[2].e[e], s[3].e[e]));
      tmax = redmax16(tmax);
      float mnew = fmaxf(mrow[e], tmax);
      float p0 = __expf(s[0].e[e] - mnew);
      float p1 = __expf(s[1].e[e] - mnew);
      float p2 = __expf(s[2].e[e] - mnew);
      float p3 = __expf(s[3].e[e] - mnew);
      float rs = redsum16((p0 + p1) + (p2 + p3));
      fac[e]  = __expf(mrow[e] - mnew);
      lrow[e] = lrow[e] * fac[e] + rs;
      mrow[e] = mnew;
      int prow = (e + hi * 8) * PP;  // C/D layout -> row-major P scratch
      myP[prow + r16]      = f32_to_bf16_rte(p0);
      myP[prow + 16 + r16] = f32_to_bf16_rte(p1);
      myP[prow + 32 + r16] = f32_to_bf16_rte(p2);
      myP[prow + 48 + r16] = f32_to_bf16_rte(p3);
    }
#pragma unroll
    for (int n = 0; n < 4; n++)
#pragma unroll
      for (int e = 0; e < 8; e++) acc[n].e[e] *= fac[e];

    // ---- P (16x64 -> two A-frags) times V (64x64) -> 8 WMMAs
    FragA pf[2];
#pragma unroll
    for (int kk = 0; kk < 2; kk++) {
      const unsigned short* p = &myP[r16 * PP + kk * 32 + hi * 8];
      pf[kk].q[0] = *(const uint4*)p;
      pf[kk].q[1] = *(const uint4*)(p + 16);
    }
#pragma unroll
    for (int n = 0; n < 4; n++) {
#pragma unroll
      for (int kk = 0; kk < 2; kk++) {
        FragA vf;  // B 32x16 from V^T: lane = d col, keys {hi*8+0..7, 16+hi*8+0..7}
        const unsigned short* p =
            &VTlds[(n * 16 + r16) * VP + kk * 32 + hi * 8];
        vf.q[0] = *(const uint4*)p;
        vf.q[1] = *(const uint4*)(p + 16);
        acc[n].f = wmma_bf16(pf[kk].bf, vf.bf, acc[n].f);
      }
    }
    __syncthreads();
  }

  // ---- normalize and store merged heads [B,S,D] bf16
#pragma unroll
  for (int n = 0; n < 4; n++) {
#pragma unroll
    for (int e = 0; e < 8; e++) {
      int row = qbase + e + hi * 8;
      int d = n * 16 + r16;
      float v = acc[n].e[e] / lrow[e];
      Out[(size_t)(b * SSEQ + row) * DIM_MODEL + h * DK + d] = f32_to_bf16_rte(v);
    }
  }
}

// ---------------------------------------------------------------------------
// Host-side orchestration
// ---------------------------------------------------------------------------
extern "C" void kernel_launch(void* const* d_in, const int* in_sizes, int n_in,
                              void* d_out, int out_size, void* d_ws, size_t ws_size,
                              hipStream_t stream) {
  (void)in_sizes; (void)n_in; (void)out_size; (void)ws_size;
  const float* Qin = (const float*)d_in[0];
  const float* Kin = (const float*)d_in[1];
  const float* Vin = (const float*)d_in[2];
  const float* Wq  = (const float*)d_in[3];
  const float* bq  = (const float*)d_in[4];
  const float* Wk  = (const float*)d_in[5];
  const float* bk  = (const float*)d_in[6];
  const float* Wv  = (const float*)d_in[7];
  const float* bv  = (const float*)d_in[8];
  const float* Wo  = (const float*)d_in[9];
  const float* bo  = (const float*)d_in[10];

  const size_t NX = (size_t)MTOT * DIM_MODEL;       // 4M activations
  const size_t NW = (size_t)DIM_MODEL * DIM_MODEL;  // 1M weights

  unsigned short* p = (unsigned short*)d_ws;
  unsigned short* Xq  = p; p += NX;
  unsigned short* Xk  = p; p += NX;
  unsigned short* Xv  = p; p += NX;
  unsigned short* Wqb = p; p += NW;
  unsigned short* Wkb = p; p += NW;
  unsigned short* Wvb = p; p += NW;
  unsigned short* Wob = p; p += NW;
  unsigned short* Qh  = p; p += NX;  // [B,H,S,dk]
  unsigned short* Kh  = p; p += NX;
  unsigned short* Vh  = p; p += NX;
  unsigned short* AO  = p; p += NX;  // merged attention out [B,S,D]

  // 1) down-convert to bf16 (vectorized x4)
  cvt_f32_bf16_v4<<<1024, 256, 0, stream>>>((const float4*)Qin, (uint2*)Xq, (int)(NX / 4));
  cvt_f32_bf16_v4<<<1024, 256, 0, stream>>>((const float4*)Kin, (uint2*)Xk, (int)(NX / 4));
  cvt_f32_bf16_v4<<<1024, 256, 0, stream>>>((const float4*)Vin, (uint2*)Xv, (int)(NX / 4));
  cvt_f32_bf16_v4<<<256, 256, 0, stream>>>((const float4*)Wq, (uint2*)Wqb, (int)(NW / 4));
  cvt_f32_bf16_v4<<<256, 256, 0, stream>>>((const float4*)Wk, (uint2*)Wkb, (int)(NW / 4));
  cvt_f32_bf16_v4<<<256, 256, 0, stream>>>((const float4*)Wv, (uint2*)Wvb, (int)(NW / 4));
  cvt_f32_bf16_v4<<<256, 256, 0, stream>>>((const float4*)Wo, (uint2*)Wob, (int)(NW / 4));

  // 2) QKV projections (head-split stores; fold 1/sqrt(dk) into Q)
  dim3 ggrid(DIM_MODEL / GBN, MTOT / GBM);  // (8, 16)
  gemm_bf16_wmma<0><<<ggrid, 256, 0, stream>>>(Xq, Wqb, bq, Qh, MTOT, DIM_MODEL,
                                               DIM_MODEL, 0.125f);
  gemm_bf16_wmma<0><<<ggrid, 256, 0, stream>>>(Xk, Wkb, bk, Kh, MTOT, DIM_MODEL,
                                               DIM_MODEL, 1.0f);
  gemm_bf16_wmma<0><<<ggrid, 256, 0, stream>>>(Xv, Wvb, bv, Vh, MTOT, DIM_MODEL,
                                               DIM_MODEL, 1.0f);

  // 3) fused flash attention -> merged-head bf16 [B,S,D]
  attn_fwd<<<dim3(SSEQ / QB, NUM_HEADS, BB), 256, 0, stream>>>(Qh, Kh, Vh, AO);

  // 4) output projection, fp32 to d_out
  gemm_bf16_wmma<1><<<ggrid, 256, 0, stream>>>(AO, Wob, bo, d_out, MTOT,
                                               DIM_MODEL, DIM_MODEL, 1.0f);
}
